// GumbelGCN_24129126269433
// MI455X (gfx1250) — matrix operand order repaired and checked by reading, not compiled, not run on hardware
//
#include <hip/hip_runtime.h>

typedef __attribute__((ext_vector_type(4)))  _Float16 v4h;
typedef __attribute__((ext_vector_type(8)))  _Float16 v8h;
typedef __attribute__((ext_vector_type(16))) _Float16 v16h;
typedef __attribute__((ext_vector_type(4)))  float    v4f;
typedef __attribute__((ext_vector_type(8)))  float    v8f;

#define N_NODES 1024
#define N_EDGES 16384
#define D_IN    128
#define ED_DIM  16
#define KTOP    8
#define NEW_E   (N_NODES * KTOP)   // 8192
#define H1_DIM  64
#define H2_DIM  64
#define OUT_DIM 40
#define NEG_INF_V (-1000000000.0f)

// ---------------------------------------------------------------- utilities
__global__ void k_zero_i32(int* p, int n) {
  int i = blockIdx.x * blockDim.x + threadIdx.x;
  if (i < n) p[i] = 0;
}
__global__ void k_fill_i32(int* p, int n, int v) {
  int i = blockIdx.x * blockDim.x + threadIdx.x;
  if (i < n) p[i] = v;
}

// ------------------------------------------------- edge scatter: adj / s_edge / first-edge map
__global__ void k_edge_scatter(const float* __restrict__ edge_attr,
                               const int*   __restrict__ edge_index,
                               const float* __restrict__ mlp_w,
                               int* adjcnt, float* s_edge, int* edge_map) {
  int e = blockIdx.x * blockDim.x + threadIdx.x;
  if (e >= N_EDGES) return;
  int s = edge_index[e];
  int d = edge_index[N_EDGES + e];
  float sc = 0.f;
#pragma unroll
  for (int k = 0; k < ED_DIM; k++)
    sc += edge_attr[e * ED_DIM + k] * mlp_w[2 * D_IN + k];  // w_e
  long off = (long)s * N_NODES + d;
  atomicAdd(&adjcnt[off], 1);
  s_edge[off] = sc;                 // .set semantics (any writer)
  atomicMin(&edge_map[off], e);     // first matching original edge (argmax of bool eq)
}

// ------------------------------------------------- per-node score vectors
__global__ void k_node_vec(const float* __restrict__ x, const float* __restrict__ mlp_w,
                           float* xw_node, float* xw_nb) {
  int i = blockIdx.x * blockDim.x + threadIdx.x;
  if (i >= N_NODES) return;
  float a = 0.f, b = 0.f;
#pragma unroll 4
  for (int k = 0; k < D_IN; k++) {
    float v = x[i * D_IN + k];
    a += v * mlp_w[k];
    b += v * mlp_w[D_IN + k];
  }
  xw_node[i] = a;
  xw_nb[i]   = b;
}

// ------------------------------------------------- fused score + softmax + gumbel + top-8
__global__ __launch_bounds__(256)
void k_row_topk(const int* __restrict__ adjcnt, const float* __restrict__ s_edge,
                const float* __restrict__ xw_node, const float* __restrict__ xw_nb,
                const float* __restrict__ mlp_b, const float* __restrict__ gumbel_u,
                const int* __restrict__ edge_map, int* new_dst, int* new_eid) {
  __shared__ float y[N_NODES];
  __shared__ float red[256];
  __shared__ int   redi[256];
  int i = blockIdx.x, t = threadIdx.x;
  float xwn = xw_node[i], bb = mlp_b[0];

  float lmax = -3.4e38f;
  for (int j = t; j < N_NODES; j += 256) {
    int c = adjcnt[i * N_NODES + j];
    float s = (c > 0) ? (xwn + xw_nb[j] + s_edge[i * N_NODES + j] + bb) : NEG_INF_V;
    y[j] = s;
    lmax = fmaxf(lmax, s);
  }
  red[t] = lmax; __syncthreads();
  for (int s = 128; s > 0; s >>= 1) { if (t < s) red[t] = fmaxf(red[t], red[t + s]); __syncthreads(); }
  float mx = red[0]; __syncthreads();

  float lsum = 0.f;
  for (int j = t; j < N_NODES; j += 256) { float ev = __expf(y[j] - mx); y[j] = ev; lsum += ev; }
  red[t] = lsum; __syncthreads();
  for (int s = 128; s > 0; s >>= 1) { if (t < s) red[t] += red[t + s]; __syncthreads(); }
  float den = red[0]; __syncthreads();

  // z + gumbel (second softmax is monotone; only top_k indices are consumed)
  for (int j = t; j < N_NODES; j += 256) {
    float u = gumbel_u[i * N_NODES + j];
    float g = -__logf(-__logf(u + 1e-20f) + 1e-20f);
    y[j] = y[j] / den + g;
  }
  __syncthreads();

  for (int k = 0; k < KTOP; k++) {
    float bv = -3.4e38f; int bi = 0;
    for (int j = t; j < N_NODES; j += 256) {
      float v = y[j];
      if (v > bv || (v == bv && j < bi)) { bv = v; bi = j; }
    }
    red[t] = bv; redi[t] = bi; __syncthreads();
    for (int s = 128; s > 0; s >>= 1) {
      if (t < s) {
        if (red[t + s] > red[t] || (red[t + s] == red[t] && redi[t + s] < redi[t])) {
          red[t] = red[t + s]; redi[t] = redi[t + s];
        }
      }
      __syncthreads();
    }
    if (t == 0) {
      int wi = redi[0];
      y[wi] = -3.4e38f;
      int eid = edge_map[i * N_NODES + wi];
      if (eid == 0x7fffffff) eid = 0;   // no original edge -> argmax(all-false)=0
      new_dst[i * KTOP + k] = wi;
      new_eid[i * KTOP + k] = eid;
    }
    __syncthreads();
  }
}

// ------------------------------------------------- WMMA fragment loaders (contiguous k order, same for A and B)
// A (16-bit 16x32): lane half=lane>>4, m=lane&15; element e -> k = e + 16*half
//  => two contiguous 8-half (16B) LDS reads per lane.
__device__ inline v16h load_a_frag(const _Float16* As, int lane) {
  int half = lane >> 4, m = lane & 15;
  v8h a0 = *(const v8h*)(As + m * 32 + 16 * half);
  v8h a1 = *(const v8h*)(As + m * 32 + 8 + 16 * half);
  return __builtin_shufflevector(a0, a1, 0, 1, 2, 3, 4, 5, 6, 7, 8, 9, 10, 11, 12, 13, 14, 15);
}
// B (16-bit 32x16): element e -> k = e + 16*half at column n=lane&15.
// Staged transposed (Bt[col][k]) => one contiguous 16-half (32B) LDS read per lane.
__device__ inline v16h load_b_frag(const _Float16* Bt, int col0, int lane) {
  int half = lane >> 4, n = lane & 15;
  return *(const v16h*)(Bt + (col0 + n) * 32 + 16 * half);
}

// ------------------------------------------------- generic WMMA GEMM: C = act(A@B + bias) [+mask]
// A: MxK f32 (M%16==0, K%32==0), B: KxN f32. block = 128 thr (4 waves), tile = 16(M) x 64(N).
template <int N, int K, bool RELU, bool MASKED>
__global__ __launch_bounds__(128)
void k_wmma_gemm(const float* __restrict__ A, const float* __restrict__ B,
                 const float* __restrict__ bias, float* __restrict__ C,
                 const int* __restrict__ mask) {
  constexpr bool NG = (N % 64) != 0;  // need column guards (only N=40)
  __shared__ __attribute__((aligned(32))) _Float16 As[16 * 32];
  __shared__ __attribute__((aligned(32))) _Float16 Bt[64 * 32];
  const int tid = threadIdx.x, lane = tid & 31, w = tid >> 5;
  const int tm = blockIdx.x * 16, tn = blockIdx.y * 64;
  const int ar = tid >> 3, ak = (tid & 7) * 4;  // A staging coords (4 floats/thread)
  v8f acc = {};
  for (int k0 = 0; k0 < K; k0 += 32) {
    // ---- stage A tile 16x32 (vector f32x4 load -> packed f16x4 LDS store)
    {
      v4f av = *(const v4f*)&A[(size_t)(tm + ar) * K + k0 + ak];
      v4h ah = {(_Float16)av[0], (_Float16)av[1], (_Float16)av[2], (_Float16)av[3]};
      *(v4h*)&As[ar * 32 + ak] = ah;
    }
    // ---- stage B tile 32x64 transposed into Bt[col][k]
#pragma unroll
    for (int i = 0; i < 4; i++) {
      int idx = tid + i * 128;
      int kk = idx >> 4, qc = idx & 15;
      int n0 = tn + qc * 4;
      if (NG) {
#pragma unroll
        for (int j = 0; j < 4; j++) {
          float v = (n0 + j < N) ? B[(size_t)(k0 + kk) * N + n0 + j] : 0.f;
          Bt[(qc * 4 + j) * 32 + kk] = (_Float16)v;
        }
      } else {
        v4f bv = *(const v4f*)&B[(size_t)(k0 + kk) * N + n0];
#pragma unroll
        for (int j = 0; j < 4; j++) Bt[(qc * 4 + j) * 32 + kk] = (_Float16)bv[j];
      }
    }
    if (k0 + 32 < K) {  // overlap next chunk fetch with this chunk's WMMA
      __builtin_prefetch(&A[(size_t)(tm + ar) * K + k0 + 32 + ak], 0, 1);
      __builtin_prefetch(&B[(size_t)(k0 + 32 + (tid >> 4)) * N + tn + (tid & 15) * 4], 0, 1);
    }
    __syncthreads();
    v16h a = load_a_frag(As, lane);
    v16h b = load_b_frag(Bt, w * 16, lane);
    acc = __builtin_amdgcn_wmma_f32_16x16x32_f16(false, a, false, b, (short)0, acc, false, false);
    __syncthreads();
  }
  int ncol = tn + w * 16 + (lane & 15);
  if (!NG || ncol < N) {
    float bv = bias[ncol];
    int mo = (lane >> 4) * 8;  // C/D layout: lanes 16-31 hold M=8..15
#pragma unroll
    for (int r = 0; r < 8; r++) {
      int row = tm + mo + r;
      float v = acc[r] + bv;
      if (RELU) v = fmaxf(v, 0.f);
      if (MASKED) { if (mask[row] == 0) v = 0.f; }
      C[(size_t)row * N + ncol] = v;
    }
  }
}

// ------------------------------------------------- GENConv edge stage (WMMA, gathered A):
// ep = edge_attr[eid] @ We + be ; m = relu(x[src]+ep)+eps ; per-dst segment max via
// int-punned float atomicMax (valid: m >= 1e-7 > 0).
template <int D>
__global__ __launch_bounds__(128)
void k_edge_m_wmma(const float* __restrict__ edge_attr, const int* __restrict__ new_eid,
                   const int* __restrict__ new_dst, const float* __restrict__ We,
                   const float* __restrict__ be, const float* __restrict__ xin,
                   float* __restrict__ m_out, int* __restrict__ mx_out) {
  __shared__ __attribute__((aligned(32))) _Float16 As[16 * 32];
  __shared__ __attribute__((aligned(32))) _Float16 Bt[64 * 32];
  const int tid = threadIdx.x, lane = tid & 31, w = tid >> 5;
  const int te = blockIdx.x * 16, tn = blockIdx.y * 64;
  // ---- stage gathered A (K=16 zero-padded to 32)
  {
    int r = tid >> 3, kq = (tid & 7) * 4;
    v4h ah = {};
    if (kq < ED_DIM) {
      int eid = new_eid[te + r];
      v4f av = *(const v4f*)&edge_attr[(size_t)eid * ED_DIM + kq];
      ah = (v4h){(_Float16)av[0], (_Float16)av[1], (_Float16)av[2], (_Float16)av[3]};
    }
    *(v4h*)&As[r * 32 + kq] = ah;
  }
  // ---- stage We transposed (rows >= ED_DIM zero)
#pragma unroll
  for (int i = 0; i < 4; i++) {
    int idx = tid + i * 128;
    int kk = idx >> 4, qc = idx & 15;
    int n0 = tn + qc * 4;  // always < D (grid.y = D/64)
    v4h bh = {};
    if (kk < ED_DIM) {
      v4f bv = *(const v4f*)&We[(size_t)kk * D + n0];
      bh = (v4h){(_Float16)bv[0], (_Float16)bv[1], (_Float16)bv[2], (_Float16)bv[3]};
    }
#pragma unroll
    for (int j = 0; j < 4; j++) Bt[(qc * 4 + j) * 32 + kk] = bh[j];
  }
  __syncthreads();
  v16h a = load_a_frag(As, lane);
  v16h b = load_b_frag(Bt, w * 16, lane);
  v8f acc = {};
  acc = __builtin_amdgcn_wmma_f32_16x16x32_f16(false, a, false, b, (short)0, acc, false, false);
  int ncol = tn + w * 16 + (lane & 15);
  float bv = be[ncol];
  int half = lane >> 4;
#pragma unroll
  for (int r = 0; r < 8; r++) {
    int ee = te + half * 8 + r;
    int node = ee >> 3;  // new_src = repeat(arange(N), 8)
    float v = xin[(size_t)node * D + ncol] + (acc[r] + bv);
    float mval = fmaxf(v, 0.f) + 1e-7f;  // GEN_EPS, strictly > 0
    m_out[(size_t)ee * D + ncol] = mval;
    int d = new_dst[ee];
    atomicMax(&mx_out[(size_t)d * D + ncol], __float_as_int(mval));
  }
}

// ------------------------------------------------- GENConv: w=exp(m - mx[dst]); den/num sums
__global__ void k_edge_w(const float* __restrict__ m, const int* __restrict__ new_dst,
                         const int* __restrict__ mx, int D, float* den, float* num) {
  int ee = blockIdx.x, f = threadIdx.x;
  if (f >= D) return;
  float mv  = m[(size_t)ee * D + f];
  int   d   = new_dst[ee];
  float mxv = __int_as_float(mx[(size_t)d * D + f]);
  float wv  = __expf(mv - mxv);
  atomicAdd(&den[(size_t)d * D + f], wv);
  atomicAdd(&num[(size_t)d * D + f], wv * mv);
}

// ------------------------------------------------- GENConv: agg + residual
__global__ void k_node_agg(const float* __restrict__ num, const float* __restrict__ den,
                           const float* __restrict__ xin, int D, float* hpre) {
  int i = blockIdx.x, f = threadIdx.x;
  if (f >= D) return;
  float dv  = den[(size_t)i * D + f];
  float agg = num[(size_t)i * D + f] / (dv > 0.f ? dv : 1.f);
  hpre[(size_t)i * D + f] = agg + xin[(size_t)i * D + f];
}

// ================================================================ launch
extern "C" void kernel_launch(void* const* d_in, const int* in_sizes, int n_in,
                              void* d_out, int out_size, void* d_ws, size_t ws_size,
                              hipStream_t stream) {
  const float* x         = (const float*)d_in[0];
  const float* edge_attr = (const float*)d_in[1];
  const float* gumbel_u  = (const float*)d_in[2];
  const float* mlp_w     = (const float*)d_in[3];
  const float* mlp_b     = (const float*)d_in[4];
  const float* c1_we     = (const float*)d_in[5];
  const float* c1_be     = (const float*)d_in[6];
  const float* c1_w1     = (const float*)d_in[7];
  const float* c1_b1     = (const float*)d_in[8];
  const float* c1_w2     = (const float*)d_in[9];
  const float* c1_b2     = (const float*)d_in[10];
  const float* c2_we     = (const float*)d_in[11];
  const float* c2_be     = (const float*)d_in[12];
  const float* c2_w1     = (const float*)d_in[13];
  const float* c2_b1     = (const float*)d_in[14];
  const float* c2_w2     = (const float*)d_in[15];
  const float* c2_b2     = (const float*)d_in[16];
  const float* fc_w      = (const float*)d_in[17];
  const float* fc_b      = (const float*)d_in[18];
  const int*   edge_index = (const int*)d_in[20];
  const int*   node_mask  = (const int*)d_in[21];
  float* out = (float*)d_out;

  const size_t NN = (size_t)N_NODES * N_NODES;
  size_t off = 0;
  char* base = (char*)d_ws;
  auto take = [&](size_t bytes) -> void* {
    void* p = base + off;
    off = (off + bytes + 255) & ~(size_t)255;
    return p;
  };
  int*   adjcnt   = (int*)  take(NN * 4);
  float* s_edge   = (float*)take(NN * 4);
  int*   edge_map = (int*)  take(NN * 4);
  float* xw_node  = (float*)take(N_NODES * 4);
  float* xw_nb    = (float*)take(N_NODES * 4);
  int*   new_dst  = (int*)  take(NEW_E * 4);
  int*   new_eid  = (int*)  take(NEW_E * 4);
  float* m1    = (float*)take((size_t)NEW_E * D_IN * 4);
  int*   mx1   = (int*)  take((size_t)N_NODES * D_IN * 4);
  float* den1  = (float*)take((size_t)N_NODES * D_IN * 4);
  float* num1  = (float*)take((size_t)N_NODES * D_IN * 4);
  float* hpre1 = (float*)take((size_t)N_NODES * D_IN * 4);
  float* t1    = (float*)take((size_t)N_NODES * 2 * D_IN * 4);
  float* h1    = (float*)take((size_t)N_NODES * H1_DIM * 4);
  float* m2    = (float*)take((size_t)NEW_E * H1_DIM * 4);
  int*   mx2   = (int*)  take((size_t)N_NODES * H1_DIM * 4);
  float* den2  = (float*)take((size_t)N_NODES * H1_DIM * 4);
  float* num2  = (float*)take((size_t)N_NODES * H1_DIM * 4);
  float* hpre2 = (float*)take((size_t)N_NODES * H1_DIM * 4);
  float* t2    = (float*)take((size_t)N_NODES * 2 * H1_DIM * 4);
  float* h2    = (float*)take((size_t)N_NODES * H2_DIM * 4);

  // zero all workspace (fresh accumulators every call -> graph-replay safe)
  int total_i32 = (int)(off / 4);
  k_zero_i32<<<(total_i32 + 255) / 256, 256, 0, stream>>>((int*)d_ws, total_i32);
  k_fill_i32<<<((int)NN + 255) / 256, 256, 0, stream>>>(edge_map, (int)NN, 0x7fffffff);

  // dense maps + node score vectors
  k_edge_scatter<<<(N_EDGES + 255) / 256, 256, 0, stream>>>(edge_attr, edge_index, mlp_w,
                                                            adjcnt, s_edge, edge_map);
  k_node_vec<<<(N_NODES + 255) / 256, 256, 0, stream>>>(x, mlp_w, xw_node, xw_nb);

  // fused score/softmax/gumbel/top-8 + edge-attr lookup
  k_row_topk<<<N_NODES, 256, 0, stream>>>(adjcnt, s_edge, xw_node, xw_nb, mlp_b, gumbel_u,
                                          edge_map, new_dst, new_eid);

  // ---------------- GENConv layer 1 (D=128)
  k_edge_m_wmma<D_IN><<<dim3(NEW_E / 16, D_IN / 64), 128, 0, stream>>>(
      edge_attr, new_eid, new_dst, c1_we, c1_be, x, m1, mx1);
  k_edge_w<<<NEW_E, D_IN, 0, stream>>>(m1, new_dst, mx1, D_IN, den1, num1);
  k_node_agg<<<N_NODES, D_IN, 0, stream>>>(num1, den1, x, D_IN, hpre1);
  k_wmma_gemm<2 * D_IN, D_IN, true, false>
      <<<dim3(N_NODES / 16, (2 * D_IN) / 64), 128, 0, stream>>>(hpre1, c1_w1, c1_b1, t1, nullptr);
  k_wmma_gemm<H1_DIM, 2 * D_IN, true, false>   // +b2 then outer relu
      <<<dim3(N_NODES / 16, 1), 128, 0, stream>>>(t1, c1_w2, c1_b2, h1, nullptr);

  // ---------------- GENConv layer 2 (D=64)
  k_edge_m_wmma<H1_DIM><<<dim3(NEW_E / 16, 1), 128, 0, stream>>>(
      edge_attr, new_eid, new_dst, c2_we, c2_be, h1, m2, mx2);
  k_edge_w<<<NEW_E, H1_DIM, 0, stream>>>(m2, new_dst, mx2, H1_DIM, den2, num2);
  k_node_agg<<<N_NODES, H1_DIM, 0, stream>>>(num2, den2, h1, H1_DIM, hpre2);
  k_wmma_gemm<2 * H1_DIM, H1_DIM, true, false>
      <<<dim3(N_NODES / 16, (2 * H1_DIM) / 64), 128, 0, stream>>>(hpre2, c2_w1, c2_b1, t2, nullptr);
  k_wmma_gemm<H2_DIM, 2 * H1_DIM, true, false>
      <<<dim3(N_NODES / 16, 1), 128, 0, stream>>>(t2, c2_w2, c2_b2, h2, nullptr);

  // ---------------- final FC + node mask -> d_out
  k_wmma_gemm<OUT_DIM, H2_DIM, false, true>
      <<<dim3(N_NODES / 16, 1), 128, 0, stream>>>(h2, fc_w, fc_b, out, node_mask);
}
